// DeLELSTM_AttnNoDecomp_11235634446894
// MI455X (gfx1250) — compile-verified, matrix operand order checked by reading, not run on hardware
//
#include <hip/hip_runtime.h>
#include <hip/hip_bf16.h>

#define D_   32
#define H_   128
#define N_   128
#define T_   256
#define B_   64
#define HEADS_ 4
#define HD_  32
#define SHORT_ 2
#define TOUT_ (T_ - 1 - (SHORT_ + 1))   // 252 predictions per batch

typedef __attribute__((ext_vector_type(16))) _Float16 v16h;
typedef __attribute__((ext_vector_type(8)))  _Float16 v8h;
typedef __attribute__((ext_vector_type(8)))  float    v8f;

__device__ __forceinline__ float sigf(float x){ return 1.0f / (1.0f + __expf(-x)); }

// ---------------------------------------------------------------------------
// WMMA fragment loaders (CDNA5 ISA 7.12.2 layouts, wave32)
// A-pattern: lane = M row (lane&15); K halves: [kg..kg+7] and [16+kg..16+kg+7]
// where kg = (lane>=16)*8.  Two contiguous 16B runs -> 2x b128 loads.
// ---------------------------------------------------------------------------
__device__ __forceinline__ v16h load_afrag(const _Float16* __restrict__ p,
                                           int stride, int lane){
  const int row = lane & 15;
  const int kg  = (lane >> 4) << 3;
  const _Float16* r = p + (size_t)row * stride + kg;
  v8h lo = *(const v8h*)(r);
  v8h hi = *(const v8h*)(r + 16);
  return __builtin_shufflevector(lo, hi, 0,1,2,3,4,5,6,7,8,9,10,11,12,13,14,15);
}

// Fragment-ready ("packed") weight load: 16 contiguous halves per lane.
__device__ __forceinline__ v16h load_pfrag(const _Float16* __restrict__ p, int lane){
  return *(const v16h*)(p + (size_t)lane * 16);
}

// ---------------------------------------------------------------------------
// Prep: fp32 -> fp16 fragment-swizzled weight mirrors, transposed mixture
// weights, zeroed recurrent state.  Runs every call (deterministic).
// ---------------------------------------------------------------------------
__global__ void k_prep(const float* __restrict__ Wj, const float* __restrict__ Wi,
                       const float* __restrict__ Wf, const float* __restrict__ Wo,
                       const float* __restrict__ qkvw, const float* __restrict__ outw,
                       const float* __restrict__ uj, const float* __restrict__ ui,
                       const float* __restrict__ uf, const float* __restrict__ uo,
                       const float* __restrict__ wj, const float* __restrict__ wi,
                       const float* __restrict__ wf, const float* __restrict__ wo,
                       _Float16* __restrict__ W16s, _Float16* __restrict__ qkvswz,
                       _Float16* __restrict__ outwswz,
                       float* __restrict__ uT, float* __restrict__ wT,
                       _Float16* __restrict__ h16a, _Float16* __restrict__ h16b,
                       float* __restrict__ cvar,
                       float* __restrict__ hmix, float* __restrict__ cmix)
{
  const long idx = (long)blockIdx.x * blockDim.x + threadIdx.x;
  const long stp = (long)gridDim.x * blockDim.x;

  // W fragment swizzle: [g][d][nt(8)][ks(4)][lane(32)][i16(16)]  (1<<21 elems)
  for (long i = idx; i < (1L << 21); i += stp){
    const int i16 = (int)(i & 15);
    const int lane = (int)((i >> 4) & 31);
    const int ks  = (int)((i >> 9) & 3);
    const int nt  = (int)((i >> 11) & 7);
    const int d   = (int)((i >> 14) & 31);
    const int g   = (int)(i >> 19);
    const int col = lane & 15, kg = (lane >> 4) << 3;
    const int k = ks * 32 + ((i16 < 8) ? (kg + i16) : (16 + kg + (i16 - 8)));
    const int n = nt * 16 + col;
    const float* s = (g == 0) ? Wj : (g == 1) ? Wi : (g == 2) ? Wf : Wo;
    W16s[i] = (_Float16)s[((long)d * H_ + k) * H_ + n];
  }
  // qkv_w transposed-B swizzle: [nt(24)][ks(4)][lane(32)][16]
  for (long i = idx; i < 24L * 4 * 32 * 16; i += stp){
    const int i16 = (int)(i & 15);
    const int lane = (int)((i >> 4) & 31);
    const int ks  = (int)((i >> 9) & 3);
    const int nt  = (int)(i >> 11);
    const int kg = (lane >> 4) << 3;
    const int k = ks * 32 + ((i16 < 8) ? (kg + i16) : (16 + kg + (i16 - 8)));
    const int nrow = nt * 16 + (lane & 15);
    qkvswz[i] = (_Float16)qkvw[(long)nrow * H_ + k];
  }
  // out_w transposed-B swizzle: [nt(8)][ks(4)][lane(32)][16]
  for (long i = idx; i < 8L * 4 * 32 * 16; i += stp){
    const int i16 = (int)(i & 15);
    const int lane = (int)((i >> 4) & 31);
    const int ks  = (int)((i >> 9) & 3);
    const int nt  = (int)(i >> 11);
    const int kg = (lane >> 4) << 3;
    const int k = ks * 32 + ((i16 < 8) ? (kg + i16) : (16 + kg + (i16 - 8)));
    const int nrow = nt * 16 + (lane & 15);
    outwswz[i] = (_Float16)outw[(long)nrow * H_ + k];
  }
  // mixture u transpose: uT[g][n][dd]
  for (long i = idx; i < 4L * 128 * 32; i += stp){
    const int dd = (int)(i & 31);
    const int n  = (int)((i >> 5) & 127);
    const int g  = (int)(i >> 12);
    const float* s = (g == 0) ? uj : (g == 1) ? ui : (g == 2) ? uf : uo;
    uT[i] = s[(long)dd * N_ + n];
  }
  // mixture w transpose: wT[g][n][m]
  for (long i = idx; i < 4L * 128 * 128; i += stp){
    const int m = (int)(i & 127);
    const int n = (int)((i >> 7) & 127);
    const int g = (int)(i >> 14);
    const float* s = (g == 0) ? wj : (g == 1) ? wi : (g == 2) ? wf : wo;
    wT[i] = s[(long)m * N_ + n];
  }
  for (long i = idx; i < (long)B_ * D_ * H_; i += stp){
    h16a[i] = (_Float16)0.0f; h16b[i] = (_Float16)0.0f; cvar[i] = 0.0f;
  }
  for (long i = idx; i < (long)B_ * N_; i += stp){ hmix[i] = 0.0f; cmix[i] = 0.0f; }
}

// ---------------------------------------------------------------------------
// Kernel A: tensor LSTM step.  grid = D_*4 blocks (d x N-quarter), 128 thr
// (4 waves, one 16-batch M-tile each).  Reads hprev, writes hnext (ping-pong)
// so the N-split across blocks is race-free.  No LDS, no barriers.
// ---------------------------------------------------------------------------
__global__ __launch_bounds__(128) void k_tlstm(
    const _Float16* __restrict__ W16s,      // [g][d][nt][ks][lane][16] fp16
    const float* __restrict__ x,            // (B,T,D)
    const float* __restrict__ Uj, const float* __restrict__ Ui,
    const float* __restrict__ Uf, const float* __restrict__ Uo,
    const float* __restrict__ Bj, const float* __restrict__ Bi,
    const float* __restrict__ Bf, const float* __restrict__ Bo,
    const _Float16* __restrict__ hprev,     // (B,D,H) fp16 h(t-1)
    _Float16* __restrict__ hnext,           // (B,D,H) fp16 h(t)
    float* __restrict__ cvar,               // (B,D,H) fp32 state
    int t)
{
  const int d     = blockIdx.x >> 2;
  const int nt0   = (blockIdx.x & 3) * 2;   // this block's 2 N-tiles
  const int wave  = threadIdx.x >> 5;
  const int lane  = threadIdx.x & 31;
  const int bbase = wave * 16;
  const int col   = lane & 15;
  const int rb    = (lane >> 4) << 3;

  // Preload A fragments: hprev[bbase.., d, 0..127]  (row stride = D*H)
  const _Float16* Ab = hprev + (size_t)bbase * (D_ * H_) + (size_t)d * H_;
  v16h afrag[4];
#pragma unroll
  for (int ks = 0; ks < 4; ++ks) afrag[ks] = load_afrag(Ab + ks * 32, D_ * H_, lane);

#pragma unroll
  for (int q = 0; q < 2; ++q){
    const int nt = nt0 + q;
    v8f acc[4];
#pragma unroll
    for (int g = 0; g < 4; ++g){
      v8f c = {0.f,0.f,0.f,0.f,0.f,0.f,0.f,0.f};
      const _Float16* Wg = W16s + ((((size_t)g * D_ + d) * 8 + nt) * 4) * (32 * 16);
#pragma unroll
      for (int ks = 0; ks < 4; ++ks){
        v16h bfr = load_pfrag(Wg + (size_t)ks * 32 * 16, lane);
        c = __builtin_amdgcn_wmma_f32_16x16x32_f16(false, afrag[ks], false, bfr,
                                                   (short)0, c, false, false);
      }
      acc[g] = c;
    }
    // Fused gate math in the C layout: lane col -> h, VGPR r -> batch row.
    const int h = nt * 16 + col;
    const float uj = Uj[d*H_+h], ui = Ui[d*H_+h], uf = Uf[d*H_+h], uo = Uo[d*H_+h];
    const float cbj = Bj[d*H_+h], cbi = Bi[d*H_+h], cbf = Bf[d*H_+h], cbo = Bo[d*H_+h];
#pragma unroll
    for (int r = 0; r < 8; ++r){
      const int b = bbase + rb + r;
      const float xv = x[(size_t)b * T_ * D_ + (size_t)t * D_ + d];
      const float jp = acc[0][r] + xv * uj + cbj;
      const float ip = acc[1][r] + xv * ui + cbi;
      const float fp = acc[2][r] + xv * uf + cbf;
      const float op = acc[3][r] + xv * uo + cbo;
      const size_t off = (size_t)b * D_ * H_ + (size_t)d * H_ + h;
      const float c0 = cvar[off];
      const float cn = c0 * sigf(fp) + sigf(ip) * tanhf(jp);
      const float hn = sigf(op) * tanhf(cn);
      cvar[off]  = cn;
      hnext[off] = (_Float16)hn;
    }
  }
}

// ---------------------------------------------------------------------------
// Kernel B: attention + mixture LSTM + prediction. grid = B_ blocks,
// 128 threads (4 waves). wave == head for the 32x32 attention stages.
// ---------------------------------------------------------------------------
__global__ __launch_bounds__(128) void k_attn_mix(
    const _Float16* __restrict__ h16,       // (B,D,H) fp16 (post tensor-LSTM)
    const _Float16* __restrict__ qkvswz,    // [nt24][ks4][lane][16] fp16
    const _Float16* __restrict__ outwswz,   // [nt8][ks4][lane][16] fp16
    const float* __restrict__ hproj,        // (N,H)
    const float* __restrict__ wp,           // (N,1)
    const float* __restrict__ bp,           // (1,)
    const float* __restrict__ x,            // (B,T,D)
    const float* __restrict__ uT,           // [g][n][dd]
    const float* __restrict__ wT,           // [g][n][m]
    const float* __restrict__ mbj, const float* __restrict__ mbi,
    const float* __restrict__ mbf, const float* __restrict__ mbo,
    float* __restrict__ hmix, float* __restrict__ cmix,
    float* __restrict__ out, int t)
{
  extern __shared__ char smem[];
  _Float16* qkv_s = (_Float16*)smem;                              // 32x384 f16
  float*    sc_s  = (float*)(smem + 32 * 384 * 2);                // 4x32x32 f32
  _Float16* a_s   = (_Float16*)((char*)sc_s + 4 * 32 * 32 * 4);   // 4x32x32 f16
  _Float16* ctx_s = a_s + 4 * 32 * 32;                            // 32x128 f16
  _Float16* vt_s  = ctx_s + 32 * 128;                             // 128x32 f16 (v^T)
  float*    z_s   = (float*)((char*)vt_s + 128 * 32 * 2);         // 32x128 f32
  float*    zm_s  = z_s + 32 * 128;                               // 128 f32
  float*    red_s = zm_s + 128;                                   // 128 f32
  float*    hm_s  = red_s + 128;                                  // 128 f32
  float*    xv_s  = hm_s + 128;                                   // 32 f32

  const int b    = blockIdx.x;
  const int tid  = threadIdx.x;
  const int wave = tid >> 5;
  const int lane = tid & 31;
  const int col  = lane & 15;
  const int rb   = (lane >> 4) << 3;

  const _Float16* hb = h16 + (size_t)b * D_ * H_;

  // Phase 1: qkv = h(32x128) @ qkvw^T -> 32x384.
  // Schedule: mt fixed per wave (wave&1), so A fragments load once per wave.
  {
    const int mt = wave & 1;
    v16h afr[4];
#pragma unroll
    for (int ks = 0; ks < 4; ++ks)
      afr[ks] = load_afrag(hb + (size_t)mt * 16 * H_ + ks * 32, H_, lane);

    for (int nt = (wave >> 1); nt < 24; nt += 2){
      v8f c = {0.f,0.f,0.f,0.f,0.f,0.f,0.f,0.f};
#pragma unroll
      for (int ks = 0; ks < 4; ++ks){
        v16h bfr = load_pfrag(qkvswz + ((size_t)nt * 4 + ks) * (32 * 16), lane);
        c = __builtin_amdgcn_wmma_f32_16x16x32_f16(false, afr[ks], false, bfr,
                                                   (short)0, c, false, false);
      }
#pragma unroll
      for (int r = 0; r < 8; ++r){
        const float v = c[r];
        qkv_s[(mt * 16 + rb + r) * 384 + nt * 16 + col] = (_Float16)v;
        if (nt >= 16)  // v-section: also store transposed for the ctx matmul
          vt_s[((nt - 16) * 16 + col) * 32 + (mt * 16 + rb + r)] = (_Float16)v;
      }
    }
  }
  __syncthreads();

  // Phase 2: scores = q @ k^T * scale   (wave == head, 2x2 16-tiles, K=32)
  {
    const int head = wave;
    const float scale = 0.17677669529663687f;   // 1/sqrt(HD=32)
#pragma unroll
    for (int it = 0; it < 2; ++it){
#pragma unroll
      for (int jt = 0; jt < 2; ++jt){
        v16h a  = load_afrag(qkv_s + (it * 16) * 384 + head * 32, 384, lane);
        v16h bk = load_afrag(qkv_s + (jt * 16) * 384 + 128 + head * 32, 384, lane);
        v8f c = {0.f,0.f,0.f,0.f,0.f,0.f,0.f,0.f};
        c = __builtin_amdgcn_wmma_f32_16x16x32_f16(false, a, false, bk,
                                                   (short)0, c, false, false);
#pragma unroll
        for (int r = 0; r < 8; ++r)
          sc_s[head * 1024 + (it * 16 + rb + r) * 32 + jt * 16 + col] = c[r] * scale;
      }
    }
  }
  __syncthreads();

  // Softmax + sparsity threshold: one row per thread (head = wave, row = lane)
  {
    float* sr = sc_s + wave * 1024 + lane * 32;
    float m = sr[0];
#pragma unroll
    for (int j = 1; j < 32; ++j) m = fmaxf(m, sr[j]);
    float e[32]; float s = 0.f;
#pragma unroll
    for (int j = 0; j < 32; ++j){ e[j] = __expf(sr[j] - m); s += e[j]; }
    const float inv = 1.0f / s;
    _Float16* ar = a_s + wave * 1024 + lane * 32;
#pragma unroll
    for (int j = 0; j < 32; ++j){
      const float av = e[j] * inv;
      ar[j] = (_Float16)((av >= 0.01f) ? av : 0.0f);
    }
  }
  __syncthreads();

  // Phase 3: ctx = a @ v  (per head; B operand from v^T via A-pattern loads)
  {
    const int head = wave;
#pragma unroll
    for (int it = 0; it < 2; ++it){
#pragma unroll
      for (int dt = 0; dt < 2; ++dt){
        v16h a  = load_afrag(a_s + head * 1024 + (it * 16) * 32, 32, lane);
        v16h bv = load_afrag(vt_s + (head * 32 + dt * 16) * 32, 32, lane);
        v8f c = {0.f,0.f,0.f,0.f,0.f,0.f,0.f,0.f};
        c = __builtin_amdgcn_wmma_f32_16x16x32_f16(false, a, false, bv,
                                                   (short)0, c, false, false);
#pragma unroll
        for (int r = 0; r < 8; ++r)
          ctx_s[(it * 16 + rb + r) * 128 + head * 32 + dt * 16 + col] = (_Float16)c[r];
      }
    }
  }
  __syncthreads();

  // Phase 4: z = ctx @ outw^T (32x128), 2 Mtiles x 8 Ntiles
  for (int tt = wave; tt < 16; tt += 4){
    const int mt = tt & 1, nt = tt >> 1;
    v8f c = {0.f,0.f,0.f,0.f,0.f,0.f,0.f,0.f};
#pragma unroll
    for (int ks = 0; ks < 4; ++ks){
      v16h a   = load_afrag(ctx_s + (size_t)mt * 16 * 128 + ks * 32, 128, lane);
      v16h bfr = load_pfrag(outwswz + ((size_t)nt * 4 + ks) * (32 * 16), lane);
      c = __builtin_amdgcn_wmma_f32_16x16x32_f16(false, a, false, bfr,
                                                 (short)0, c, false, false);
    }
#pragma unroll
    for (int r = 0; r < 8; ++r)
      z_s[(mt * 16 + rb + r) * 128 + nt * 16 + col] = c[r];
  }
  __syncthreads();

  // Phase 5a: zmean over the D axis; stage hmix row and x_t into LDS
  {
    float s = 0.f;
#pragma unroll
    for (int i = 0; i < 32; ++i) s += z_s[i * 128 + tid];
    zm_s[tid] = s * (1.0f / 32.0f);
    hm_s[tid] = hmix[b * N_ + tid];
    if (tid < 32) xv_s[tid] = x[(size_t)b * T_ * D_ + (size_t)t * D_ + tid];
  }
  __syncthreads();

  // Phase 5b: h_hat, mixture LSTM, h_mix select, prediction
  {
    const int n = tid;
    float hh = 0.f;
#pragma unroll 8
    for (int h = 0; h < 128; ++h) hh += zm_s[h] * hproj[(size_t)n * 128 + h];

    const float* ujT = uT + 0 * 4096 + n * 32;
    const float* uiT = uT + 1 * 4096 + n * 32;
    const float* ufT = uT + 2 * 4096 + n * 32;
    const float* uoT = uT + 3 * 4096 + n * 32;
    const float* wjT = wT + 0 * 16384 + n * 128;
    const float* wiT = wT + 1 * 16384 + n * 128;
    const float* wfT = wT + 2 * 16384 + n * 128;
    const float* woT = wT + 3 * 16384 + n * 128;

    float gj = mbj[n], gi = mbi[n], gf = mbf[n], go = mbo[n];
#pragma unroll 8
    for (int dd = 0; dd < 32; ++dd){
      const float xv = xv_s[dd];
      gj += xv * ujT[dd]; gi += xv * uiT[dd];
      gf += xv * ufT[dd]; go += xv * uoT[dd];
    }
#pragma unroll 8
    for (int m = 0; m < 128; ++m){
      const float hm = hm_s[m];
      gj += hm * wjT[m]; gi += hm * wiT[m];
      gf += hm * wfT[m]; go += hm * woT[m];
    }
    const float cm  = cmix[b * N_ + n];
    const float cn  = sigf(gf) * cm + sigf(gi) * tanhf(gj);
    const float hml = sigf(go) * tanhf(cn);
    const float hnew = (t > SHORT_) ? hh : hml;

    red_s[tid] = hh * wp[n];
    __syncthreads();
    cmix[b * N_ + n] = cn;
    hmix[b * N_ + n] = hnew;

    for (int s2 = 64; s2 > 0; s2 >>= 1){
      if (tid < s2) red_s[tid] += red_s[tid + s2];
      __syncthreads();
    }
    if (tid == 0 && t >= SHORT_ + 1)
      out[(size_t)b * TOUT_ + (t - (SHORT_ + 1))] = red_s[0] + bp[0];
  }
}

// ---------------------------------------------------------------------------
extern "C" void kernel_launch(void* const* d_in, const int* in_sizes, int n_in,
                              void* d_out, int out_size, void* d_ws, size_t ws_size,
                              hipStream_t stream)
{
  const float* x    = (const float*)d_in[0];
  const float* Uj   = (const float*)d_in[1];
  const float* Ui_  = (const float*)d_in[2];
  const float* Uf   = (const float*)d_in[3];
  const float* Uo   = (const float*)d_in[4];
  const float* Wj   = (const float*)d_in[5];
  const float* Wi   = (const float*)d_in[6];
  const float* Wf   = (const float*)d_in[7];
  const float* Wo   = (const float*)d_in[8];
  const float* Bj   = (const float*)d_in[9];
  const float* Bi   = (const float*)d_in[10];
  const float* Bf   = (const float*)d_in[11];
  const float* Bo   = (const float*)d_in[12];
  const float* uj   = (const float*)d_in[13];
  const float* ui   = (const float*)d_in[14];
  const float* uf   = (const float*)d_in[15];
  const float* uo   = (const float*)d_in[16];
  const float* wj   = (const float*)d_in[17];
  const float* wi   = (const float*)d_in[18];
  const float* wf   = (const float*)d_in[19];
  const float* wo   = (const float*)d_in[20];
  const float* bj   = (const float*)d_in[21];
  const float* bi   = (const float*)d_in[22];
  const float* bf   = (const float*)d_in[23];
  const float* bo   = (const float*)d_in[24];
  const float* qkvw = (const float*)d_in[25];
  const float* outw = (const float*)d_in[26];
  const float* hproj= (const float*)d_in[27];
  const float* wp   = (const float*)d_in[28];
  const float* bp   = (const float*)d_in[29];
  float* out = (float*)d_out;

  // Workspace carve-up (256B aligned)
  char* ws = (char*)d_ws;
  size_t off = 0;
  auto carve = [&](size_t bytes) -> char* {
    char* p = ws + off;
    off += (bytes + 255) & ~(size_t)255;
    return p;
  };
  _Float16* h16a    = (_Float16*)carve((size_t)B_ * D_ * H_ * 2);      // 512 KB
  _Float16* h16b    = (_Float16*)carve((size_t)B_ * D_ * H_ * 2);      // 512 KB
  float*    cvar    = (float*)   carve((size_t)B_ * D_ * H_ * 4);      // 1 MB
  float*    hmix    = (float*)   carve((size_t)B_ * N_ * 4);           // 32 KB
  float*    cmix    = (float*)   carve((size_t)B_ * N_ * 4);           // 32 KB
  _Float16* W16s    = (_Float16*)carve((size_t)(1 << 21) * 2);         // 4 MB swizzled
  _Float16* qkvswz  = (_Float16*)carve((size_t)24 * 4 * 32 * 16 * 2);  // 96 KB
  _Float16* outwswz = (_Float16*)carve((size_t)8 * 4 * 32 * 16 * 2);   // 32 KB
  float*    uT      = (float*)   carve((size_t)4 * 128 * 32 * 4);      // 64 KB
  float*    wT      = (float*)   carve((size_t)4 * 128 * 128 * 4);     // 256 KB
  (void)ws_size; (void)in_sizes; (void)n_in; (void)out_size;

  k_prep<<<512, 256, 0, stream>>>(Wj, Wi, Wf, Wo, qkvw, outw,
                                  uj, ui, uf, uo, wj, wi, wf, wo,
                                  W16s, qkvswz, outwswz, uT, wT,
                                  h16a, h16b, cvar, hmix, cmix);

  const size_t smemB = (size_t)32*384*2   // qkv
                     + (size_t)4*32*32*4  // scores
                     + (size_t)4*32*32*2  // a (f16)
                     + (size_t)32*128*2   // ctx (f16)
                     + (size_t)128*32*2   // v^T (f16)
                     + (size_t)32*128*4   // z
                     + 128*4 + 128*4 + 128*4 + 32*4; // zmean + reduce + hmix + x_t

  for (int t = 0; t < T_ - 1; ++t){
    const _Float16* hp = (t & 1) ? h16b : h16a;   // h(t-1)
    _Float16*       hn = (t & 1) ? h16a : h16b;   // h(t)
    k_tlstm<<<D_ * 4, 128, 0, stream>>>(W16s, x, Uj, Ui_, Uf, Uo, Bj, Bi, Bf, Bo,
                                        hp, hn, cvar, t);
    k_attn_mix<<<B_, 128, smemB, stream>>>(hn, qkvswz, outwswz, hproj, wp, bp, x,
                                           uT, wT, bj, bi, bf, bo,
                                           hmix, cmix, out, t);
  }
}
